// GraphNetwork_28741921145146
// MI455X (gfx1250) — matrix operand order repaired and checked by reading, not compiled, not run on hardware
//
#include <hip/hip_runtime.h>
#include <hip/hip_bf16.h>

typedef __attribute__((ext_vector_type(16))) _Float16 v16h;
typedef __attribute__((ext_vector_type(8)))  float    v8f;

// ---------------------------------------------------------------------------
// WMMA wrapper: D = A(16x32 f16) * B(32x16 f16) + C(16x16 f32)
// ---------------------------------------------------------------------------
__device__ __forceinline__ v8f wmma16(v16h a, v16h b, v8f c) {
  return __builtin_amdgcn_wmma_f32_16x16x32_f16(
      /*neg_a=*/false, a, /*neg_b=*/false, b,
      /*c_mod=*/(short)0, c, /*reuse_a=*/false, /*reuse_b=*/false);
}

__device__ __forceinline__ v8f splat8(float v) {
  v8f c;
#pragma unroll
  for (int r = 0; r < 8; ++r) c[r] = v;
  return c;
}

// float atomic max via int-max / uint-min trick (buffer initialized to -inf)
__device__ __forceinline__ void atomicMaxF(float* addr, float v) {
  if (v >= 0.0f) atomicMax((int*)addr, __float_as_int(v));
  else           atomicMin((unsigned int*)addr, __float_as_uint(v));
}

// ---------------------------------------------------------------------------
// A-fragment (16x32 f16) from fp32 matrix M[*,64].
// lane = 16*h + m holds row m; slot j -> K = kbase + 8h + (j<8 ? j : 16+(j&7))
// Two contiguous 8-float runs -> vectorizes to global_load_b128 pairs.
// ---------------------------------------------------------------------------
__device__ __forceinline__ v16h load_afrag64(const float* __restrict__ M, long row,
                                             int kbase, int h, bool fixneg) {
  const float* p = M + row * 64 + kbase + 8 * h;
  v16h a;
#pragma unroll
  for (int j = 0; j < 16; ++j) {
    int off = ((j >> 3) << 4) + (j & 7);
    float v = p[off];
    if (fixneg) v = (v > -1e30f) ? v : 0.0f;   // -inf (empty segment) -> 0
    a[j] = (_Float16)v;
  }
  return a;
}

// B-fragment (32x16 f16) from LDS-staged transposed weights Wt[n][kstride].
// lane = 16*h + n holds column n; slot j -> K = kbase + 8h + (j<8 ? j : 16+(j&7))
__device__ __forceinline__ v16h load_bfrag_lds(const _Float16* Wt, int kstride,
                                               int n, int h, int kbase) {
  const _Float16* p = Wt + n * kstride + kbase + 8 * h;
  v16h b;
#pragma unroll
  for (int j = 0; j < 16; ++j) b[j] = p[((j >> 3) << 4) + (j & 7)];
  return b;
}

// ---------------------------------------------------------------------------
// Kernel 1: fused edge MLP  h = relu([x_j | pos_j-pos_i] @ Wl1 + bl1) @ Wl2 + bl2
// then atomic-max aggregate into agg[dst].  16 edges per wave, 8 waves/block.
// ---------------------------------------------------------------------------
__global__ __launch_bounds__(256)
void k_edge_mlp(const float* __restrict__ x, const float* __restrict__ pos,
                const int* __restrict__ src, const int* __restrict__ dst,
                const float* __restrict__ Wl1, const float* __restrict__ bl1,
                const float* __restrict__ Wl2, const float* __restrict__ bl2,
                float* __restrict__ agg, int E) {
  __shared__ _Float16 sW1[64 * 32];     // Wl1^T zero-padded: [n][k0..31], k>=6 -> 0
  __shared__ _Float16 sW2[64 * 64];     // Wl2^T: [n][k]
  __shared__ _Float16 sH[8][16 * 64];   // per-wave 16x64 f16 tile, col-major

  const int tid  = threadIdx.x;
  const int lane = tid & 31;
  const int wv   = tid >> 5;
  const int m    = lane & 15;
  const int h    = lane >> 4;
  const long ebase = ((long)blockIdx.x * 8 + wv) * 16;

  // ---- stage weights to LDS: coalesced linear reads, transposed f16 writes
#pragma unroll
  for (int i = tid; i < 64 * 32; i += 256) {
    int k = i >> 6, n = i & 63;                       // i = k*64 + n (linear)
    float w = Wl1[(k < 6 ? k : 5) * 64 + n];
    sW1[n * 32 + k] = (k < 6) ? (_Float16)w : (_Float16)0.0f;
  }
#pragma unroll
  for (int i = tid; i < 64 * 64; i += 256) {
    int k = i >> 6, n = i & 63;
    sW2[n * 64 + k] = (_Float16)Wl2[i];               // Wl2[k*64+n] == Wl2[i]
  }

  // ---- layer-1 A fragment: K=6 zero-padded to 32.  K>=8 lives in lanes 16-31 => zero.
  v16h a1;
#pragma unroll
  for (int j = 0; j < 16; ++j) a1[j] = (_Float16)0.0f;
  {
    long e = ebase + m; if (e >= E) e = E - 1;
    if (h == 0) {
      int s = src[e], d = dst[e];
      a1[0] = (_Float16)x[s * 3 + 0];
      a1[1] = (_Float16)x[s * 3 + 1];
      a1[2] = (_Float16)x[s * 3 + 2];
      a1[3] = (_Float16)(pos[s * 3 + 0] - pos[d * 3 + 0]);
      a1[4] = (_Float16)(pos[s * 3 + 1] - pos[d * 3 + 1]);
      a1[5] = (_Float16)(pos[s * 3 + 2] - pos[d * 3 + 2]);
    }
  }

  // hoist dst indices for the scatter (8 per lane, reused across all 4 n-tiles)
  int dvals[8];
#pragma unroll
  for (int r = 0; r < 8; ++r) {
    long e = ebase + r + 8 * h;
    dvals[r] = dst[e < E ? e : (E - 1)];
  }

  __syncthreads();   // weights staged

  // ---- layer 1: 4 N-tiles, relu, stash f16 tile to LDS (col-major) for re-layout
#pragma unroll
  for (int t = 0; t < 4; ++t) {
    const int n = 16 * t + m;
    v16h b = load_bfrag_lds(sW1, 32, n, h, 0);
    v8f c = splat8(bl1[n]);
    c = wmma16(a1, b, c);
    _Float16* p = &sH[wv][n * 16 + 8 * h];            // contiguous 8 halfs
#pragma unroll
    for (int r = 0; r < 8; ++r) {
      float v = c[r];
      p[r] = (_Float16)(v > 0.0f ? v : 0.0f);
    }
  }
  __syncthreads();

  // ---- re-read as layer-2 A fragments (K = 0..63)
  v16h a2lo, a2hi;
#pragma unroll
  for (int j = 0; j < 16; ++j) {
    int kl = ((j >> 3) << 4) + 8 * h + (j & 7);       // 0..31
    a2lo[j] = sH[wv][kl * 16 + m];
    a2hi[j] = sH[wv][(32 + kl) * 16 + m];
  }

  // ---- layer 2 + atomic-max scatter by dst
#pragma unroll
  for (int t = 0; t < 4; ++t) {
    const int n = 16 * t + m;
    v16h b0 = load_bfrag_lds(sW2, 64, n, h, 0);
    v16h b1 = load_bfrag_lds(sW2, 64, n, h, 32);
    v8f c = splat8(bl2[n]);
    c = wmma16(a2lo, b0, c);
    c = wmma16(a2hi, b1, c);
#pragma unroll
    for (int r = 0; r < 8; ++r) {
      long e = ebase + r + 8 * h;
      if (e < E) atomicMaxF(&agg[(long)dvals[r] * 64 + n], c[r]);
    }
  }
}

// ---------------------------------------------------------------------------
// Kernel 2: x1 = leaky(agg @ Wg + bg) * mask1      (agg -inf fixed to 0 on load)
// ---------------------------------------------------------------------------
__global__ __launch_bounds__(256)
void k_x1(const float* __restrict__ agg, const float* __restrict__ Wg,
          const float* __restrict__ bg, const float* __restrict__ mask1,
          float* __restrict__ x1, int N) {
  __shared__ _Float16 sWg[64 * 64];
  const int tid = threadIdx.x;
  const int lane = tid & 31, wv = tid >> 5;
  const int m = lane & 15, h = lane >> 4;
  const long nbase = ((long)blockIdx.x * 8 + wv) * 16;
  long row = nbase + m; if (row >= N) row = N - 1;

#pragma unroll
  for (int i = tid; i < 64 * 64; i += 256) {
    int k = i >> 6, n = i & 63;
    sWg[n * 64 + k] = (_Float16)Wg[i];
  }

  v16h alo = load_afrag64(agg, row, 0,  h, true);
  v16h ahi = load_afrag64(agg, row, 32, h, true);
  __syncthreads();

#pragma unroll
  for (int t = 0; t < 4; ++t) {
    const int n = 16 * t + m;
    v8f c = splat8(bg[n]);
    c = wmma16(alo, load_bfrag_lds(sWg, 64, n, h, 0),  c);
    c = wmma16(ahi, load_bfrag_lds(sWg, 64, n, h, 32), c);
#pragma unroll
    for (int r = 0; r < 8; ++r) {
      long rr = nbase + r + 8 * h;
      if (rr < N) {
        float v = c[r];
        v = (v >= 0.0f) ? v : 0.01f * v;
        x1[rr * 64 + n] = v * mask1[rr * 64 + n];
      }
    }
  }
}

// ---------------------------------------------------------------------------
// Kernel 3/5: segment-sum  s[dst] += xin[src]  (one thread = one edge x 16 ch)
// ---------------------------------------------------------------------------
__global__ __launch_bounds__(256)
void k_scatter_add(const int* __restrict__ src, const int* __restrict__ dst,
                   const float* __restrict__ xin, float* __restrict__ sout, int E) {
  long tid = (long)blockIdx.x * blockDim.x + threadIdx.x;
  long e = tid >> 2;
  if (e >= E) return;
  int c16 = (int)(tid & 3) * 16;
  int s = src[e], d = dst[e];
  const float4* vp = (const float4*)(xin + (long)s * 64 + c16);
  float4 v0 = vp[0], v1 = vp[1], v2 = vp[2], v3 = vp[3];
  float* op = sout + (long)d * 64 + c16;
  atomicAdd(op + 0,  v0.x); atomicAdd(op + 1,  v0.y);
  atomicAdd(op + 2,  v0.z); atomicAdd(op + 3,  v0.w);
  atomicAdd(op + 4,  v1.x); atomicAdd(op + 5,  v1.y);
  atomicAdd(op + 6,  v1.z); atomicAdd(op + 7,  v1.w);
  atomicAdd(op + 8,  v2.x); atomicAdd(op + 9,  v2.y);
  atomicAdd(op + 10, v2.z); atomicAdd(op + 11, v2.w);
  atomicAdd(op + 12, v3.x); atomicAdd(op + 13, v3.y);
  atomicAdd(op + 14, v3.z); atomicAdd(op + 15, v3.w);
}

// ---------------------------------------------------------------------------
// Kernel 4: x2 = leaky(s1 @ W2r + x1 @ W2s + b2) * mask2
// ---------------------------------------------------------------------------
__global__ __launch_bounds__(256)
void k_x2(const float* __restrict__ s1, const float* __restrict__ x1,
          const float* __restrict__ W2r, const float* __restrict__ W2s,
          const float* __restrict__ b2, const float* __restrict__ mask2,
          float* __restrict__ x2, int N) {
  __shared__ _Float16 sWr[64 * 64];
  __shared__ _Float16 sWs[64 * 64];
  const int tid = threadIdx.x;
  const int lane = tid & 31, wv = tid >> 5;
  const int m = lane & 15, h = lane >> 4;
  const long nbase = ((long)blockIdx.x * 8 + wv) * 16;
  long row = nbase + m; if (row >= N) row = N - 1;

#pragma unroll
  for (int i = tid; i < 64 * 64; i += 256) {
    int k = i >> 6, n = i & 63;
    sWr[n * 64 + k] = (_Float16)W2r[i];
    sWs[n * 64 + k] = (_Float16)W2s[i];
  }

  v16h aslo = load_afrag64(s1, row, 0,  h, false);
  v16h ashi = load_afrag64(s1, row, 32, h, false);
  v16h axlo = load_afrag64(x1, row, 0,  h, false);
  v16h axhi = load_afrag64(x1, row, 32, h, false);
  __syncthreads();

#pragma unroll
  for (int t = 0; t < 4; ++t) {
    const int n = 16 * t + m;
    v8f c = splat8(b2[n]);
    c = wmma16(aslo, load_bfrag_lds(sWr, 64, n, h, 0),  c);
    c = wmma16(ashi, load_bfrag_lds(sWr, 64, n, h, 32), c);
    c = wmma16(axlo, load_bfrag_lds(sWs, 64, n, h, 0),  c);
    c = wmma16(axhi, load_bfrag_lds(sWs, 64, n, h, 32), c);
#pragma unroll
    for (int r = 0; r < 8; ++r) {
      long rr = nbase + r + 8 * h;
      if (rr < N) {
        float v = c[r];
        v = (v >= 0.0f) ? v : 0.01f * v;
        x2[rr * 64 + n] = v * mask2[rr * 64 + n];
      }
    }
  }
}

// ---------------------------------------------------------------------------
// Kernel 6: x3 = s2 @ W3r + x2 @ W3s + b3 ; atomic-max pool into out[batch]
// ---------------------------------------------------------------------------
__global__ __launch_bounds__(256)
void k_x3(const float* __restrict__ s2, const float* __restrict__ x2,
          const float* __restrict__ W3r, const float* __restrict__ W3s,
          const float* __restrict__ b3, const int* __restrict__ batch,
          float* __restrict__ out, int N) {
  __shared__ _Float16 sWr[32 * 64];   // W3r^T: [n][k], n<32, k<64
  __shared__ _Float16 sWs[32 * 64];
  const int tid = threadIdx.x;
  const int lane = tid & 31, wv = tid >> 5;
  const int m = lane & 15, h = lane >> 4;
  const long nbase = ((long)blockIdx.x * 8 + wv) * 16;
  long row = nbase + m; if (row >= N) row = N - 1;

#pragma unroll
  for (int i = tid; i < 64 * 32; i += 256) {
    int k = i >> 5, n = i & 31;                       // i = k*32 + n (linear)
    sWr[n * 64 + k] = (_Float16)W3r[i];
    sWs[n * 64 + k] = (_Float16)W3s[i];
  }

  v16h aslo = load_afrag64(s2, row, 0,  h, false);
  v16h ashi = load_afrag64(s2, row, 32, h, false);
  v16h axlo = load_afrag64(x2, row, 0,  h, false);
  v16h axhi = load_afrag64(x2, row, 32, h, false);

  int gvals[8];
#pragma unroll
  for (int r = 0; r < 8; ++r) {
    long rr = nbase + r + 8 * h;
    gvals[r] = batch[rr < N ? rr : (N - 1)];
  }
  __syncthreads();

#pragma unroll
  for (int t = 0; t < 2; ++t) {                       // OUT = 32 -> two N-tiles
    const int n = 16 * t + m;
    v8f c = splat8(b3[n]);
    c = wmma16(aslo, load_bfrag_lds(sWr, 64, n, h, 0),  c);
    c = wmma16(ashi, load_bfrag_lds(sWr, 64, n, h, 32), c);
    c = wmma16(axlo, load_bfrag_lds(sWs, 64, n, h, 0),  c);
    c = wmma16(axhi, load_bfrag_lds(sWs, 64, n, h, 32), c);
#pragma unroll
    for (int r = 0; r < 8; ++r) {
      long rr = nbase + r + 8 * h;
      if (rr < N) atomicMaxF(&out[(long)gvals[r] * 32 + n], c[r]);
    }
  }
}

// ---------------------------------------------------------------------------
// Utility kernels
// ---------------------------------------------------------------------------
__global__ void k_fill(float* __restrict__ p, long n, float v) {
  long i = (long)blockIdx.x * blockDim.x + threadIdx.x;
  if (i < n) p[i] = v;
}
__global__ void k_fixup(float* __restrict__ p, int n) {
  int i = blockIdx.x * blockDim.x + threadIdx.x;
  if (i < n) { float v = p[i]; if (!(v > -1e30f) || !(v < 1e30f)) p[i] = 0.0f; }
}

// ---------------------------------------------------------------------------
extern "C" void kernel_launch(void* const* d_in, const int* in_sizes, int n_in,
                              void* d_out, int out_size, void* d_ws, size_t ws_size,
                              hipStream_t stream) {
  const float* x     = (const float*)d_in[0];
  const float* pos   = (const float*)d_in[1];
  const int*   ei    = (const int*)  d_in[2];
  const int*   batch = (const int*)  d_in[3];
  const float* mask1 = (const float*)d_in[4];
  const float* mask2 = (const float*)d_in[5];
  const float* Wl1   = (const float*)d_in[6];
  const float* bl1   = (const float*)d_in[7];
  const float* Wl2   = (const float*)d_in[8];
  const float* bl2   = (const float*)d_in[9];
  const float* Wg    = (const float*)d_in[10];
  const float* bg    = (const float*)d_in[11];
  const float* W2r   = (const float*)d_in[12];
  const float* W2s   = (const float*)d_in[13];
  const float* b2    = (const float*)d_in[14];
  const float* W3r   = (const float*)d_in[15];
  const float* W3s   = (const float*)d_in[16];
  const float* b3    = (const float*)d_in[17];

  const int N = in_sizes[0] / 3;
  const int E = in_sizes[2] / 2;
  const int* src = ei;
  const int* dst = ei + E;

  float* ws  = (float*)d_ws;
  const long N64 = (long)N * 64;
  float* agg = ws;              // 16 MB
  float* x1  = ws + N64;        // 16 MB
  float* s1  = ws + 2 * N64;    // 16 MB
  float* x2  = ws + 3 * N64;    // 16 MB
  float* s2  = agg;             // reuse: agg is dead after k_x1
  float* out = (float*)d_out;

  const float NEG_INF = -__builtin_inff();
  const int fillBlk = (int)((N64 + 255) / 256);

  k_fill<<<fillBlk, 256, 0, stream>>>(agg, N64, NEG_INF);
  k_fill<<<fillBlk, 256, 0, stream>>>(s1, N64, 0.0f);
  k_fill<<<(out_size + 255) / 256, 256, 0, stream>>>(out, (long)out_size, NEG_INF);

  k_edge_mlp<<<(E + 127) / 128, 256, 0, stream>>>(x, pos, src, dst,
                                                  Wl1, bl1, Wl2, bl2, agg, E);
  k_x1<<<(N + 127) / 128, 256, 0, stream>>>(agg, Wg, bg, mask1, x1, N);
  k_fill<<<fillBlk, 256, 0, stream>>>(s2, N64, 0.0f);
  k_scatter_add<<<(int)(((long)E * 4 + 255) / 256), 256, 0, stream>>>(src, dst, x1, s1, E);
  k_x2<<<(N + 127) / 128, 256, 0, stream>>>(s1, x1, W2r, W2s, b2, mask2, x2, N);
  k_scatter_add<<<(int)(((long)E * 4 + 255) / 256), 256, 0, stream>>>(src, dst, x2, s2, E);
  k_x3<<<(N + 127) / 128, 256, 0, stream>>>(s2, x2, W3r, W3s, b3, batch, out, N);
  k_fixup<<<(out_size + 255) / 256, 256, 0, stream>>>(out, out_size);
}
